// Embed_RNN_14422500180468
// MI455X (gfx1250) — compile-verified, matrix operand order checked by reading, not run on hardware
//
#include <hip/hip_runtime.h>

// Problem constants (from reference)
#define EMB   32
#define HID   256
#define KDIM  288         // HID + EMB concatenated A-matrix columns
#define BSZ   64
#define TSEQ  2048
#define MTILE 16          // batch rows per workgroup
#define NTHR  512         // 16 waves
#define NSLOT 18          // per-thread LDS B-tile slots

typedef __attribute__((ext_vector_type(16))) __bf16 v16bf;
typedef __attribute__((ext_vector_type(2)))  __bf16 v2bf;
typedef __attribute__((ext_vector_type(8)))  float  v8f;

__device__ __forceinline__ v8f wmma_bf16(v16bf a, v16bf b, v8f c) {
    return __builtin_amdgcn_wmma_f32_16x16x32_bf16(false, a, false, b, (short)0, c, false, false);
}

__device__ __forceinline__ v16bf load_cvt16(const float* p) {
    v16bf r;
#pragma unroll
    for (int i = 0; i < 16; ++i) r[i] = (__bf16)p[i];
    return r;
}

__device__ __forceinline__ float sigmoid_fast(float x) {
    return 1.0f / (1.0f + __expf(-x));
}
__device__ __forceinline__ float tanh_fast(float x) {
    return 1.0f - 2.0f / (__expf(2.0f * x) + 1.0f);
}

// Exchange with lane ^ 8 (SWAPX8): group-of-32 swizzle, xor=8, and=0x1F.
__device__ __forceinline__ float swz_xor8(float v) {
    int i = __builtin_bit_cast(int, v);
    i = __builtin_amdgcn_ds_swizzle(i, 0x201F);
    return __builtin_bit_cast(float, i);
}

__global__ __launch_bounds__(NTHR, 1)
void lstm_recurrent_kernel(const int* __restrict__ x,
                           const int* __restrict__ lengths,
                           const float* __restrict__ emb,
                           const float* __restrict__ W_ih,
                           const float* __restrict__ W_hh,
                           const float* __restrict__ b_ih,
                           const float* __restrict__ b_hh,
                           float* __restrict__ out) {
    // Per-thread interleaved B-tile slots: one base register, immediate offsets.
    __shared__ __align__(32) __bf16 sh_w[NTHR * NSLOT * 16];    // 288KB
    // Unified A-matrix buffer: cols 0..255 = h (bf16), cols 256..287 = xe
    __shared__ __align__(32) __bf16 sh_hx[2][MTILE * KDIM];     // 2 x 9KB

    const int tid  = threadIdx.x;
    const int b0   = blockIdx.x * MTILE;
    const int wv   = tid >> 5;          // wave 0..15 -> hidden dims 16*wv..16*wv+15
    const int lane = tid & 31;
    const int lm   = lane & 15;
    const int kh   = lane >> 4;         // K half of A/B fragments
    const int l8   = lm & 7;
    const int hi8  = (lm >> 3) & 1;     // 0: lane's gates {i,g}; 1: {f,o}
    const int hdA  = wv * 16 + l8;      // first hidden dim (tiles 0,2)
    const int hdB  = hdA + 8;           // second hidden dim (tiles 1,3)
    const int rbase = hi8 * 4 + kh * 8; // first of 4 rows this lane updates

    // Gate rows for the 4 N-tiles: 0:[i|f]@hdA 1:[i|f]@hdB 2:[g|o]@hdA 3:[g|o]@hdB
    const int nA0 = (0 + hi8) * HID + hdA;
    const int nB0 = (0 + hi8) * HID + hdB;
    const int nA1 = (2 + hi8) * HID + hdA;
    const int nB1 = (2 + hi8) * HID + hdB;

    // ---- register-resident W_hh tiles: k0..3 x 4 cols + k4 x 2 cols (144 VGPRs) ----
    v16bf R0[4], R1[4], R2[4], R3[4];
#pragma unroll
    for (int k = 0; k < 4; ++k) {
        R0[k] = load_cvt16(W_hh + (size_t)nA0 * HID + k * 32 + kh * 16);
        R1[k] = load_cvt16(W_hh + (size_t)nB0 * HID + k * 32 + kh * 16);
        R2[k] = load_cvt16(W_hh + (size_t)nA1 * HID + k * 32 + kh * 16);
        R3[k] = load_cvt16(W_hh + (size_t)nB1 * HID + k * 32 + kh * 16);
    }
    v16bf R4a = load_cvt16(W_hh + (size_t)nA0 * HID + 4 * 32 + kh * 16);
    v16bf R4b = load_cvt16(W_hh + (size_t)nB0 * HID + 4 * 32 + kh * 16);

    // ---- LDS-resident B-tiles (per-thread slots, consumed by same thread) ----
    // 0..3: W_ih cols 0..3 (K-tile 8) | 4,5: W_hh k4 cols 2,3 | 6..9: k5 | 10..13: k6 | 14..17: k7
    __bf16* wrow = &sh_w[(size_t)tid * (NSLOT * 16)];
    *(v16bf*)&wrow[0 * 16]  = load_cvt16(W_ih + (size_t)nA0 * EMB + kh * 16);
    *(v16bf*)&wrow[1 * 16]  = load_cvt16(W_ih + (size_t)nB0 * EMB + kh * 16);
    *(v16bf*)&wrow[2 * 16]  = load_cvt16(W_ih + (size_t)nA1 * EMB + kh * 16);
    *(v16bf*)&wrow[3 * 16]  = load_cvt16(W_ih + (size_t)nB1 * EMB + kh * 16);
    *(v16bf*)&wrow[4 * 16]  = load_cvt16(W_hh + (size_t)nA1 * HID + 4 * 32 + kh * 16);
    *(v16bf*)&wrow[5 * 16]  = load_cvt16(W_hh + (size_t)nB1 * HID + 4 * 32 + kh * 16);
#pragma unroll
    for (int k = 5; k < 8; ++k) {
        const int s = 6 + (k - 5) * 4;
        *(v16bf*)&wrow[(s + 0) * 16] = load_cvt16(W_hh + (size_t)nA0 * HID + k * 32 + kh * 16);
        *(v16bf*)&wrow[(s + 1) * 16] = load_cvt16(W_hh + (size_t)nB0 * HID + k * 32 + kh * 16);
        *(v16bf*)&wrow[(s + 2) * 16] = load_cvt16(W_hh + (size_t)nA1 * HID + k * 32 + kh * 16);
        *(v16bf*)&wrow[(s + 3) * 16] = load_cvt16(W_hh + (size_t)nB1 * HID + k * 32 + kh * 16);
    }

    // Biases packed as bf16 pairs (2 VGPRs total)
    v2bf biasP01, biasP23;
    biasP01[0] = (__bf16)(b_ih[nA0] + b_hh[nA0]);
    biasP01[1] = (__bf16)(b_ih[nB0] + b_hh[nB0]);
    biasP23[0] = (__bf16)(b_ih[nA1] + b_hh[nA1]);
    biasP23[1] = (__bf16)(b_ih[nB1] + b_hh[nB1]);

    // Lengths packed 2x u16 per VGPR (lengths <= 2048)
    const int len01 = (lengths[b0 + rbase + 0] & 0xffff) |
                      (lengths[b0 + rbase + 1] << 16);
    const int len23 = (lengths[b0 + rbase + 2] & 0xffff) |
                      (lengths[b0 + rbase + 3] << 16);

    // zero the h-region of buffer 0 (xe-region is written before first use)
    for (int i = tid; i < MTILE * KDIM / 2; i += NTHR) ((int*)sh_hx[0])[i] = 0;

    float cA[4] = {0.f, 0.f, 0.f, 0.f};
    float cB[4] = {0.f, 0.f, 0.f, 0.f};

    // ---- software-pipelined embedding gather, packed in one VGPR ----
    // pk = next-token (low 16) | current xe value as bf16 bits (high 16)
    const int ge   = tid & 31;                       // gather emb dim
    const int gxe  = (tid >> 5) * KDIM + HID + ge;   // xe cell for this thread
    const int xoff = (b0 + (tid >> 5)) * TSEQ;       // 32-bit index off uniform base
    int pk;
    {
        const float xv0 = emb[(size_t)x[xoff] * EMB + ge];
        const short  xb = __builtin_bit_cast(short, (__bf16)xv0);
        pk = (x[xoff + 1] & 0xffff) | (((int)xb & 0xffff) << 16);
    }
    __syncthreads();   // LDS weight slots + h0 ready

    int p = 0;
    for (int t = 0; t < TSEQ; ++t) {
        // ---- publish xe for step t (bf16 bits prefetched last step) ----
        sh_hx[p][gxe] = __builtin_bit_cast(__bf16, (short)((unsigned)pk >> 16));
        __syncthreads();   // xe + h of buffer p visible

        // ---- WMMA: gates = bias + [h|xe] @ [W_hh|W_ih]^T ----
        const float bias0 = (float)biasP01[0], bias1 = (float)biasP01[1];
        const float bias2 = (float)biasP23[0], bias3 = (float)biasP23[1];
        v8f acc0, acc1, acc2, acc3;
#pragma unroll
        for (int r = 0; r < 8; ++r) {
            acc0[r] = bias0; acc1[r] = bias1; acc2[r] = bias2; acc3[r] = bias3;
        }

        const __bf16* arow = &sh_hx[p][lm * KDIM + kh * 16];
#pragma unroll
        for (int k = 0; k < 4; ++k) {     // register-resident K-tiles
            v16bf A = *(const v16bf*)&arow[k * 32];
            acc0 = wmma_bf16(A, R0[k], acc0);
            acc1 = wmma_bf16(A, R1[k], acc1);
            acc2 = wmma_bf16(A, R2[k], acc2);
            acc3 = wmma_bf16(A, R3[k], acc3);
            __builtin_amdgcn_sched_barrier(0);
        }
        {
            v16bf A = *(const v16bf*)&arow[4 * 32];
            acc0 = wmma_bf16(A, R4a, acc0);
            acc1 = wmma_bf16(A, R4b, acc1);
            acc2 = wmma_bf16(A, *(const v16bf*)&wrow[4 * 16], acc2);
            acc3 = wmma_bf16(A, *(const v16bf*)&wrow[5 * 16], acc3);
            __builtin_amdgcn_sched_barrier(0);
        }
#pragma unroll
        for (int k = 5; k < 8; ++k) {     // LDS-resident K-tiles
            const int s = 6 + (k - 5) * 4;
            v16bf A = *(const v16bf*)&arow[k * 32];
            acc0 = wmma_bf16(A, *(const v16bf*)&wrow[(s + 0) * 16], acc0);
            acc1 = wmma_bf16(A, *(const v16bf*)&wrow[(s + 1) * 16], acc1);
            acc2 = wmma_bf16(A, *(const v16bf*)&wrow[(s + 2) * 16], acc2);
            acc3 = wmma_bf16(A, *(const v16bf*)&wrow[(s + 3) * 16], acc3);
            __builtin_amdgcn_sched_barrier(0);
        }
        {                                 // K-tile 8 = xe @ W_ih
            v16bf A = *(const v16bf*)&arow[8 * 32];
            acc0 = wmma_bf16(A, *(const v16bf*)&wrow[0 * 16], acc0);
            acc1 = wmma_bf16(A, *(const v16bf*)&wrow[1 * 16], acc1);
            acc2 = wmma_bf16(A, *(const v16bf*)&wrow[2 * 16], acc2);
            acc3 = wmma_bf16(A, *(const v16bf*)&wrow[3 * 16], acc3);
            __builtin_amdgcn_sched_barrier(0);
        }

        // ---- issue next-step gather early; latency hidden by phase C ----
        {
            const int tk = pk & 0xffff;
            float xnf = 0.0f;
            if (t + 1 < TSEQ) xnf = emb[(size_t)tk * EMB + ge];
            int ntok = 0;
            if (t + 2 < TSEQ) ntok = x[xoff + t + 2];
            const short xb = __builtin_bit_cast(short, (__bf16)xnf);
            pk = (ntok & 0xffff) | (((int)xb & 0xffff) << 16);
        }

        // ---- in-wave activations; update 8 (row, hd) cells ----
        int lcur[4];
        lcur[0] = len01 & 0xffff; lcur[1] = (int)((unsigned)len01 >> 16);
        lcur[2] = len23 & 0xffff; lcur[3] = (int)((unsigned)len23 >> 16);
#pragma unroll
        for (int rr = 0; rr < 4; ++rr) {
            const bool act = t < lcur[rr];
            // hdA: gates from acc0 (i|f) and acc2 (g|o)
            {
                const float a0l = acc0[rr], a0h = acc0[rr + 4];
                const float a1l = acc2[rr], a1h = acc2[rr + 4];
                const float s0l = swz_xor8(a0l), s0h = swz_xor8(a0h);
                const float s1l = swz_xor8(a1l), s1h = swz_xor8(a1h);
                const float own0 = hi8 ? a0h : a0l;
                const float oth0 = hi8 ? s0h : s0l;
                const float own1 = hi8 ? a1h : a1l;
                const float oth1 = hi8 ? s1h : s1l;
                const float iv = hi8 ? oth0 : own0;
                const float fv = hi8 ? own0 : oth0;
                const float gv = hi8 ? oth1 : own1;
                const float ov = hi8 ? own1 : oth1;
                const float ig = sigmoid_fast(iv);
                const float fg = sigmoid_fast(fv);
                const float gg = tanh_fast(gv);
                const float og = sigmoid_fast(ov);
                const float cn = fg * cA[rr] + ig * gg;
                const float hn = og * tanh_fast(cn);
                const int idx = (rbase + rr) * KDIM + hdA;
                const __bf16 holdb = sh_hx[p][idx];        // frozen-row carry
                sh_hx[p ^ 1][idx] = act ? (__bf16)hn : holdb;
                cA[rr] = act ? cn : cA[rr];
            }
            // hdB: gates from acc1 (i|f) and acc3 (g|o)
            {
                const float a0l = acc1[rr], a0h = acc1[rr + 4];
                const float a1l = acc3[rr], a1h = acc3[rr + 4];
                const float s0l = swz_xor8(a0l), s0h = swz_xor8(a0h);
                const float s1l = swz_xor8(a1l), s1h = swz_xor8(a1h);
                const float own0 = hi8 ? a0h : a0l;
                const float oth0 = hi8 ? s0h : s0l;
                const float own1 = hi8 ? a1h : a1l;
                const float oth1 = hi8 ? s1h : s1l;
                const float iv = hi8 ? oth0 : own0;
                const float fv = hi8 ? own0 : oth0;
                const float gv = hi8 ? oth1 : own1;
                const float ov = hi8 ? own1 : oth1;
                const float ig = sigmoid_fast(iv);
                const float fg = sigmoid_fast(fv);
                const float gg = tanh_fast(gv);
                const float og = sigmoid_fast(ov);
                const float cn = fg * cB[rr] + ig * gg;
                const float hn = og * tanh_fast(cn);
                const int idx = (rbase + rr) * KDIM + hdB;
                const __bf16 holdb = sh_hx[p][idx];
                sh_hx[p ^ 1][idx] = act ? (__bf16)hn : holdb;
                cB[rr] = act ? cn : cB[rr];
            }
        }
        p ^= 1;
    }

    // ---- final h (masked-carried, in last-written buffer) -> out[B, 1, HID] ----
#pragma unroll
    for (int rr = 0; rr < 4; ++rr) {
        out[(size_t)(b0 + rbase + rr) * HID + hdA] =
            (float)sh_hx[p][(rbase + rr) * KDIM + hdA];
        out[(size_t)(b0 + rbase + rr) * HID + hdB] =
            (float)sh_hx[p][(rbase + rr) * KDIM + hdB];
    }
}

extern "C" void kernel_launch(void* const* d_in, const int* in_sizes, int n_in,
                              void* d_out, int out_size, void* d_ws, size_t ws_size,
                              hipStream_t stream) {
    const int*   x       = (const int*)d_in[0];
    const int*   lengths = (const int*)d_in[1];
    const float* emb     = (const float*)d_in[2];
    const float* W_ih    = (const float*)d_in[3];
    const float* W_hh    = (const float*)d_in[4];
    const float* b_ih    = (const float*)d_in[5];
    const float* b_hh    = (const float*)d_in[6];
    float* out = (float*)d_out;

    (void)in_sizes; (void)n_in; (void)out_size; (void)d_ws; (void)ws_size;

    lstm_recurrent_kernel<<<dim3(BSZ / MTILE), dim3(NTHR), 0, stream>>>(
        x, lengths, emb, W_ih, W_hh, b_ih, b_hh, out);
}
